// MultiHeadAttention_14791867367611
// MI455X (gfx1250) — compile-verified, matrix operand order checked by reading, not run on hardware
//
#include <hip/hip_runtime.h>
#include <hip/hip_bf16.h>
#include <stdint.h>

#define B_    2
#define T_    2048
#define D_    2048
#define H_    16
#define HKV_  4
#define HD_   128
#define DKV_  512
#define MROWS (B_*T_)   // 4096

typedef __attribute__((ext_vector_type(16))) __bf16        v16bf;
typedef __attribute__((ext_vector_type(8)))  float         v8f;
typedef __attribute__((ext_vector_type(4)))  unsigned int  v4u;

union AFrag { unsigned int u[8]; v4u q[2]; v16bf v; };

__device__ __forceinline__ unsigned short f2bf(float f) {
  union { float f; unsigned int u; } v; v.f = f;
  unsigned int r = v.u + 0x7FFFu + ((v.u >> 16) & 1u);
  return (unsigned short)(r >> 16);
}

__device__ __forceinline__ v8f vzero8() {
  v8f z = {0.f,0.f,0.f,0.f,0.f,0.f,0.f,0.f};
  return z;
}

// ---------------- conversion kernels ----------------

__global__ void k_f32_to_bf16(const float* __restrict__ in,
                              unsigned short* __restrict__ out, int n) {
  int i = blockIdx.x * blockDim.x + threadIdx.x;
  if (i < n) out[i] = f2bf(in[i]);
}

// in: K x N (row major f32) -> out: N x K (row major bf16)  == B^T for A*B
__global__ void k_transpose_bf16(const float* __restrict__ in,
                                 unsigned short* __restrict__ out, int K, int N) {
  int i = blockIdx.x * blockDim.x + threadIdx.x;
  if (i < K * N) {
    int k = i / N, n = i % N;
    out[(size_t)n * K + k] = f2bf(in[i]);
  }
}

// ---------------- bf16 WMMA GEMM: C[M,N] = A[M,K] * Bt[N,K]^T ----------------
// block = 256 thr = 8 waves; block tile 128(M) x 64(N); wave w -> rows 16w..16w+15

__global__ __launch_bounds__(256) void k_gemm_bf16(
    const unsigned short* __restrict__ A,
    const unsigned short* __restrict__ Bt,
    float* __restrict__ C, int M, int N, int K)
{
  const int lane = threadIdx.x & 31;
  const int wave = threadIdx.x >> 5;
  const int g    = lane >> 4;    // half-wave select
  const int n16  = lane & 15;
  const int m0   = blockIdx.x * 128 + wave * 16;
  const int n0   = blockIdx.y * 64;
  const int Kd   = K >> 1;       // dwords (bf16 pairs) per row

  const unsigned int* A32 = (const unsigned int*)A;
  const unsigned int* B32 = (const unsigned int*)Bt;
  const unsigned int* arow  = A32 + (size_t)(m0 + n16) * Kd;
  const unsigned int* brow0 = B32 + (size_t)(n0 + n16) * Kd;
  const unsigned int* brow1 = brow0 + (size_t)16 * Kd;
  const unsigned int* brow2 = brow1 + (size_t)16 * Kd;
  const unsigned int* brow3 = brow2 + (size_t)16 * Kd;

  v8f acc0 = vzero8(), acc1 = vzero8(), acc2 = vzero8(), acc3 = vzero8();

  for (int kc = 0; kc < K; kc += 32) {
    const int db = (kc >> 1) + g * 4;
    AFrag a, b0, b1, b2, b3;
    a.q[0]  = *(const v4u*)(arow  + db);  a.q[1]  = *(const v4u*)(arow  + db + 8);
    b0.q[0] = *(const v4u*)(brow0 + db);  b0.q[1] = *(const v4u*)(brow0 + db + 8);
    b1.q[0] = *(const v4u*)(brow1 + db);  b1.q[1] = *(const v4u*)(brow1 + db + 8);
    b2.q[0] = *(const v4u*)(brow2 + db);  b2.q[1] = *(const v4u*)(brow2 + db + 8);
    b3.q[0] = *(const v4u*)(brow3 + db);  b3.q[1] = *(const v4u*)(brow3 + db + 8);
    acc0 = __builtin_amdgcn_wmma_f32_16x16x32_bf16(false, a.v, false, b0.v, (short)0, acc0, false, false);
    acc1 = __builtin_amdgcn_wmma_f32_16x16x32_bf16(false, a.v, false, b1.v, (short)0, acc1, false, false);
    acc2 = __builtin_amdgcn_wmma_f32_16x16x32_bf16(false, a.v, false, b2.v, (short)0, acc2, false, false);
    acc3 = __builtin_amdgcn_wmma_f32_16x16x32_bf16(false, a.v, false, b3.v, (short)0, acc3, false, false);
  }

  const size_t crow = (size_t)(m0 + 8 * g) * N + n0 + n16;
#pragma unroll
  for (int r = 0; r < 8; ++r) {
    C[crow + (size_t)r * N +  0] = acc0[r];
    C[crow + (size_t)r * N + 16] = acc1[r];
    C[crow + (size_t)r * N + 32] = acc2[r];
    C[crow + (size_t)r * N + 48] = acc3[r];
  }
}

// ---------------- RoPE + pack kernels ----------------

__global__ void k_rope_pack_q(const float* __restrict__ Qf,
                              unsigned short* __restrict__ qb) {
  int i = blockIdx.x * blockDim.x + threadIdx.x;
  if (i >= MROWS * D_) return;
  int d = i % D_;  int row = i / D_;
  int t = row % T_; int b = row / T_;
  int h = d >> 7;  int hi = d & 127; int j = hi & 63;
  float inv = __powf(10000.f, -(float)(2 * j) * (1.0f / 128.f));
  float ang = (float)t * inv;
  float s, c; __sincosf(ang, &s, &c);
  float x0 = Qf[i];
  float xr = (hi < 64) ? -Qf[i + 64] : Qf[i - 64];
  float val = (x0 * c + xr * s) * 0.08838834764831845f; // hd^-1/2 folded into Q
  qb[((size_t)(b * H_ + h) * T_ + t) * HD_ + hi] = f2bf(val);
}

__global__ void k_rope_pack_k(const float* __restrict__ Kf,
                              unsigned short* __restrict__ kb) {
  int i = blockIdx.x * blockDim.x + threadIdx.x;
  if (i >= MROWS * DKV_) return;
  int d = i % DKV_; int row = i / DKV_;
  int t = row % T_; int b = row / T_;
  int kvh = d >> 7; int hi = d & 127; int j = hi & 63;
  float inv = __powf(10000.f, -(float)(2 * j) * (1.0f / 128.f));
  float ang = (float)t * inv;
  float s, c; __sincosf(ang, &s, &c);
  float x0 = Kf[i];
  float xr = (hi < 64) ? -Kf[i + 64] : Kf[i - 64];
  kb[((size_t)(b * HKV_ + kvh) * T_ + t) * HD_ + hi] = f2bf(x0 * c + xr * s);
}

// V -> transposed bf16 (B, Hkv, hd, T)
__global__ void k_pack_vt(const float* __restrict__ Vf,
                          unsigned short* __restrict__ vt) {
  int i = blockIdx.x * blockDim.x + threadIdx.x;
  if (i >= MROWS * DKV_) return;
  int d = i % DKV_; int row = i / DKV_;
  int t = row % T_; int b = row / T_;
  int kvh = d >> 7; int hi = d & 127;
  vt[((size_t)(b * HKV_ + kvh) * HD_ + hi) * T_ + t] = f2bf(Vf[i]);
}

// ---------------- causal flash attention (one wave per 16-query tile) --------

__global__ __launch_bounds__(256) void k_attn(
    const unsigned short* __restrict__ qb,   // (B,H,T,hd)   bf16
    const unsigned short* __restrict__ kb,   // (B,Hkv,T,hd) bf16
    const unsigned short* __restrict__ vt,   // (B,Hkv,hd,T) bf16
    unsigned short* __restrict__ ob)         // (B*T, D)     bf16
{
  __shared__ __align__(16) unsigned int pbuf[8][16 * 16]; // per-wave 16x32 bf16 P tile
  const int lane = threadIdx.x & 31;
  const int wave = threadIdx.x >> 5;
  const int g    = lane >> 4;
  const int n16  = lane & 15;
  const int TQ   = T_ / 16;

  int tile = blockIdx.x * 8 + wave;       // 0 .. B*H*TQ-1
  int b    = tile / (H_ * TQ);
  int rem  = tile % (H_ * TQ);
  int h    = rem / TQ;
  int q0   = (rem % TQ) * 16;
  int kvh  = h >> 2;                      // H/Hkv = 4 (repeat_interleave)

  const unsigned int* q32 = (const unsigned int*)qb;
  const unsigned int* k32 = (const unsigned int*)kb;
  const unsigned int* v32 = (const unsigned int*)vt;
  const unsigned int* qrow   = q32 + ((size_t)(b * H_ + h) * T_ + q0 + n16) * (HD_ / 2);
  const unsigned int* kbasep = k32 + (size_t)(b * HKV_ + kvh) * T_ * (HD_ / 2);
  const unsigned int* vbasep = v32 + (size_t)(b * HKV_ + kvh) * HD_ * (T_ / 2);

  AFrag qf[4];
#pragma unroll
  for (int c = 0; c < 4; ++c) {
    qf[c].q[0] = *(const v4u*)(qrow + c * 16 + g * 4);
    qf[c].q[1] = *(const v4u*)(qrow + c * 16 + 8 + g * 4);
  }

  v8f o[8];
#pragma unroll
  for (int t = 0; t < 8; ++t) o[t] = vzero8();
  float mrow[8], lrow[8];
#pragma unroll
  for (int r = 0; r < 8; ++r) { mrow[r] = -3.0e38f; lrow[r] = 0.f; }

  const int nkeys = q0 + 16;
  for (int jj = 0; jj < nkeys; jj += 32) {
    int kr0 = jj + n16;      if (kr0 > T_ - 1) kr0 = T_ - 1;
    int kr1 = jj + 16 + n16; if (kr1 > T_ - 1) kr1 = T_ - 1;
    const unsigned int* krow0 = kbasep + (size_t)kr0 * (HD_ / 2);
    const unsigned int* krow1 = kbasep + (size_t)kr1 * (HD_ / 2);

    v8f s0 = vzero8(), s1 = vzero8();
#pragma unroll
    for (int c = 0; c < 4; ++c) {
      AFrag kf0, kf1;
      kf0.q[0] = *(const v4u*)(krow0 + c * 16 + g * 4);
      kf0.q[1] = *(const v4u*)(krow0 + c * 16 + 8 + g * 4);
      kf1.q[0] = *(const v4u*)(krow1 + c * 16 + g * 4);
      kf1.q[1] = *(const v4u*)(krow1 + c * 16 + 8 + g * 4);
      s0 = __builtin_amdgcn_wmma_f32_16x16x32_bf16(false, qf[c].v, false, kf0.v, (short)0, s0, false, false);
      s1 = __builtin_amdgcn_wmma_f32_16x16x32_bf16(false, qf[c].v, false, kf1.v, (short)0, s1, false, false);
    }

    unsigned short* ps = (unsigned short*)pbuf[wave];
    float alpha[8];
#pragma unroll
    for (int r = 0; r < 8; ++r) {
      int qi = q0 + 8 * g + r;
      float v0 = (jj + n16      <= qi) ? s0[r] : -3.0e38f;
      float v1 = (jj + 16 + n16 <= qi) ? s1[r] : -3.0e38f;
      float mx = fmaxf(v0, v1);
      mx = fmaxf(mx, __shfl_xor(mx, 1, 32));
      mx = fmaxf(mx, __shfl_xor(mx, 2, 32));
      mx = fmaxf(mx, __shfl_xor(mx, 4, 32));
      mx = fmaxf(mx, __shfl_xor(mx, 8, 32));
      float nm = fmaxf(mrow[r], mx);
      float al = __expf(mrow[r] - nm);
      float p0 = __expf(v0 - nm);
      float p1 = __expf(v1 - nm);
      float rs = p0 + p1;
      rs += __shfl_xor(rs, 1, 32);
      rs += __shfl_xor(rs, 2, 32);
      rs += __shfl_xor(rs, 4, 32);
      rs += __shfl_xor(rs, 8, 32);
      lrow[r] = lrow[r] * al + rs;
      mrow[r] = nm;
      alpha[r] = al;
      ps[(8 * g + r) * 32 + n16]      = f2bf(p0);   // C-layout -> LDS (row, key)
      ps[(8 * g + r) * 32 + 16 + n16] = f2bf(p1);
    }
#pragma unroll
    for (int t = 0; t < 8; ++t) {
#pragma unroll
      for (int r = 0; r < 8; ++r) o[t][r] *= alpha[r];
    }

    __builtin_amdgcn_wave_barrier();
    asm volatile("s_wait_dscnt 0x0" ::: "memory");

    // read P back in A-fragment layout (row = lane%16, key pattern per half-wave)
    AFrag pf;
    const unsigned int* prow = &pbuf[wave][n16 * 16];
#pragma unroll
    for (int v = 0; v < 4; ++v) {
      pf.u[v]     = prow[g * 4 + v];
      pf.u[4 + v] = prow[8 + g * 4 + v];
    }
    __builtin_amdgcn_wave_barrier();

    const int jd = jj >> 1;
#pragma unroll
    for (int t = 0; t < 8; ++t) {
      const unsigned int* vrow = vbasep + (size_t)(16 * t + n16) * (T_ / 2);
      AFrag vf;
      vf.q[0] = *(const v4u*)(vrow + jd + g * 4);
      vf.q[1] = *(const v4u*)(vrow + jd + 8 + g * 4);
      o[t] = __builtin_amdgcn_wmma_f32_16x16x32_bf16(false, pf.v, false, vf.v, (short)0, o[t], false, false);
    }
  }

#pragma unroll
  for (int t = 0; t < 8; ++t) {
#pragma unroll
    for (int r = 0; r < 8; ++r) {
      float val = o[t][r] / lrow[r];
      ob[((size_t)(b * T_ + q0 + 8 * g + r)) * D_ + h * HD_ + 16 * t + n16] = f2bf(val);
    }
  }
}

// ---------------- launch ----------------

extern "C" void kernel_launch(void* const* d_in, const int* in_sizes, int n_in,
                              void* d_out, int out_size, void* d_ws, size_t ws_size,
                              hipStream_t stream) {
  (void)in_sizes; (void)n_in; (void)out_size; (void)ws_size;
  const float* x  = (const float*)d_in[0];
  const float* Wq = (const float*)d_in[1];
  const float* Wk = (const float*)d_in[2];
  const float* Wv = (const float*)d_in[3];
  const float* Wo = (const float*)d_in[4];
  float* out = (float*)d_out;

  char* ws = (char*)d_ws;
  size_t off = 0;
  auto alloc = [&](size_t bytes) {
    size_t o = off; off += (bytes + 255) & ~(size_t)255; return o;
  };
  unsigned short* xb  = (unsigned short*)(ws + alloc((size_t)MROWS * D_   * 2));
  unsigned short* wqT = (unsigned short*)(ws + alloc((size_t)D_ * D_     * 2));
  unsigned short* wkT = (unsigned short*)(ws + alloc((size_t)D_ * DKV_   * 2));
  unsigned short* wvT = (unsigned short*)(ws + alloc((size_t)D_ * DKV_   * 2));
  unsigned short* woT = (unsigned short*)(ws + alloc((size_t)D_ * D_     * 2));
  float* Qf = (float*)(ws + alloc((size_t)MROWS * D_   * 4));
  float* Kf = (float*)(ws + alloc((size_t)MROWS * DKV_ * 4));
  float* Vf = (float*)(ws + alloc((size_t)MROWS * DKV_ * 4));
  unsigned short* kb = (unsigned short*)(ws + alloc((size_t)B_ * HKV_ * T_ * HD_ * 2));
  unsigned short* vt = (unsigned short*)(ws + alloc((size_t)B_ * HKV_ * HD_ * T_ * 2 + 256));
  // aliases (safe by kernel ordering): qb over Kf+Vf (both dead after packs),
  // attention output over Qf (dead after q-pack).
  unsigned short* qb    = (unsigned short*)Kf;
  unsigned short* attnb = (unsigned short*)Qf;

  const int thr = 256;
  { int n = MROWS * D_; k_f32_to_bf16<<<(n + thr - 1) / thr, thr, 0, stream>>>(x, xb, n); }
  { int n = D_ * D_;    k_transpose_bf16<<<(n + thr - 1) / thr, thr, 0, stream>>>(Wq, wqT, D_, D_); }
  { int n = D_ * DKV_;  k_transpose_bf16<<<(n + thr - 1) / thr, thr, 0, stream>>>(Wk, wkT, D_, DKV_); }
  { int n = D_ * DKV_;  k_transpose_bf16<<<(n + thr - 1) / thr, thr, 0, stream>>>(Wv, wvT, D_, DKV_); }
  { int n = D_ * D_;    k_transpose_bf16<<<(n + thr - 1) / thr, thr, 0, stream>>>(Wo, woT, D_, D_); }

  k_gemm_bf16<<<dim3(MROWS / 128, D_   / 64), 256, 0, stream>>>(xb, wqT, Qf, MROWS, D_,   D_);
  k_gemm_bf16<<<dim3(MROWS / 128, DKV_ / 64), 256, 0, stream>>>(xb, wkT, Kf, MROWS, DKV_, D_);
  k_gemm_bf16<<<dim3(MROWS / 128, DKV_ / 64), 256, 0, stream>>>(xb, wvT, Vf, MROWS, DKV_, D_);

  { int n = MROWS * DKV_; k_rope_pack_k<<<(n + thr - 1) / thr, thr, 0, stream>>>(Kf, kb); }
  { int n = MROWS * DKV_; k_pack_vt   <<<(n + thr - 1) / thr, thr, 0, stream>>>(Vf, vt); }
  { int n = MROWS * D_;   k_rope_pack_q<<<(n + thr - 1) / thr, thr, 0, stream>>>(Qf, qb); }

  k_attn<<<(B_ * H_ * (T_ / 16)) / 8, 256, 0, stream>>>(qb, kb, vt, attnb);

  k_gemm_bf16<<<dim3(MROWS / 128, D_ / 64), 256, 0, stream>>>(attnb, woT, out, MROWS, D_, D_);
}
